// BaiChuanAttention_11476152615174
// MI455X (gfx1250) — matrix operand order repaired and checked by reading, not compiled, not run on hardware
//
#include <hip/hip_runtime.h>

// ---------------------------------------------------------------------------
// BaiChuan attention for MI455X (gfx1250): bf16 WMMA GEMMs + flash attention,
// with async global->LDS staging (ASYNCcnt) and double-buffered GEMM tiles.
// ---------------------------------------------------------------------------

typedef __bf16 v16bf __attribute__((ext_vector_type(16)));
typedef float  v8f   __attribute__((ext_vector_type(8)));
typedef unsigned int u32x4 __attribute__((ext_vector_type(4)));
typedef unsigned int u32x2 __attribute__((ext_vector_type(2)));
typedef float f32x4 __attribute__((ext_vector_type(4)));

#define DEVINLINE __device__ __forceinline__

constexpr int Bb = 2, Ss = 2048, Hh = 4096, NHh = 32, HDd = 128;

#if defined(__AMDGCN__) && __has_builtin(__builtin_amdgcn_global_load_async_to_lds_b128) && \
    __has_builtin(__builtin_amdgcn_s_wait_asynccnt)
#define HAVE_ASYNC_LDS 1
typedef int v4i_vs __attribute__((vector_size(16)));
typedef __attribute__((address_space(1))) v4i_vs as1_v4i;   // global
typedef __attribute__((address_space(3))) v4i_vs as3_v4i;   // LDS
#else
#define HAVE_ASYNC_LDS 0
#endif

union FragAB {
  v16bf v;
  u32x4 q[2];
};

DEVINLINE unsigned short f2bf(float f) {
  unsigned int u = __float_as_uint(f);
  u += 0x7FFFu + ((u >> 16) & 1u);      // round-to-nearest-even
  return (unsigned short)(u >> 16);
}
DEVINLINE float bf2f(unsigned short h) {
  return __uint_as_float(((unsigned int)h) << 16);
}

// Async 16B global->LDS copy (no VGPR round-trip); sync fallback otherwise.
DEVINLINE void async_copy_b128(const unsigned short* gsrc, unsigned short* ldsdst) {
#if HAVE_ASYNC_LDS
  __builtin_amdgcn_global_load_async_to_lds_b128((as1_v4i*)gsrc, (as3_v4i*)ldsdst,
                                                 /*offset=*/0, /*cpol=*/0);
#else
  *(u32x4*)ldsdst = *(const u32x4*)gsrc;
#endif
}
DEVINLINE void async_wait0() {
#if HAVE_ASYNC_LDS
  __builtin_amdgcn_s_wait_asynccnt(0);
#endif
}

// A fragment (16x32 bf16, MxK): lane<16 -> M=lane, K = {0..7, 16..23};
// lane>=16 -> M=lane-16, K = {8..15, 24..31}. src points at [row0][k0], ld in elems.
DEVINLINE void load_frag_a(FragAB& f, const unsigned short* src, int ld, int lane) {
  const int half = lane >> 4;
  const int m = lane & 15;
  const unsigned short* p = src + (size_t)m * ld + half * 8;
  f.q[0] = *(const u32x4*)(p);
  f.q[1] = *(const u32x4*)(p + 16);
}
// B fragment (32x16 bf16, KxN), operand stored row-major as [N][K]:
// lane<16 -> N=lane, K = {0..7, 16..23}; lane>=16 -> N=lane-16, K = {8..15, 24..31}.
DEVINLINE void load_frag_b(FragAB& f, const unsigned short* src, int ld, int lane) {
  const int half = lane >> 4;
  const int n = lane & 15;
  const unsigned short* p = src + (size_t)n * ld + half * 8;
  f.q[0] = *(const u32x4*)(p);
  f.q[1] = *(const u32x4*)(p + 16);
}

DEVINLINE v8f wmma_bf16(const FragAB& a, const FragAB& b, v8f c) {
  return __builtin_amdgcn_wmma_f32_16x16x32_bf16(
      /*neg_a=*/false, a.v, /*neg_b=*/false, b.v,
      /*c_mod=*/(short)0, c, /*reuse_a=*/false, /*reuse_b=*/false);
}

// ---------------------------------------------------------------------------
// fp32 -> bf16 bulk convert (hidden_states), 4 elems/thread.
// ---------------------------------------------------------------------------
__global__ __launch_bounds__(256) void cvt_kernel(const float* __restrict__ in,
                                                  unsigned short* __restrict__ out) {
  size_t i = (size_t)blockIdx.x * blockDim.x + threadIdx.x;
  f32x4 v = *(const f32x4*)(in + i * 4);
  unsigned short o[4] = {f2bf(v.x), f2bf(v.y), f2bf(v.z), f2bf(v.w)};
  *(u32x2*)(out + i * 4) = *(const u32x2*)o;
}

// ---------------------------------------------------------------------------
// GEMM: C[M][N] = A[M][K](bf16) * W[N][K]^T(fp32->bf16 in staging).
// 128x128x32 workgroup tile, 8 waves (2M x 4N), wave tile 64x32.
// A tiles: async global->LDS, double buffered; W tiles: VALU-convert staging.
// ---------------------------------------------------------------------------
DEVINLINE void stage_a_async(const unsigned short* Abf, unsigned short (*Al)[40],
                             int m0, int K, int k0, int tid) {
#pragma unroll
  for (int i = 0; i < 2; ++i) {
    int c = tid + i * 256;            // 512 chunks of 8 bf16
    int row = c >> 2;                 // 0..127
    int col8 = (c & 3) << 3;          // 0,8,16,24
    async_copy_b128(Abf + (size_t)(m0 + row) * K + k0 + col8, &Al[row][col8]);
  }
}
DEVINLINE void stage_b_convert(const float* Wg, unsigned short (*Bl)[40],
                               int n0, int K, int k0, int tid) {
#pragma unroll
  for (int i = 0; i < 4; ++i) {
    int idx = tid + i * 256;          // 1024 chunks of 4 fp32
    int row = idx >> 3;               // 0..127
    int c4  = (idx & 7) << 2;         // 0,4,..,28
    f32x4 wv = *(const f32x4*)(Wg + (size_t)(n0 + row) * K + k0 + c4);
    unsigned short* bd = &Bl[row][c4];
    bd[0] = f2bf(wv.x); bd[1] = f2bf(wv.y); bd[2] = f2bf(wv.z); bd[3] = f2bf(wv.w);
  }
}

template <bool SCATTER_QKV>
__global__ __launch_bounds__(256) void gemm_wmma(
    const unsigned short* __restrict__ Abf, const float* __restrict__ Wg,
    float* __restrict__ Cout,
    unsigned short* __restrict__ Qo, unsigned short* __restrict__ Ko,
    unsigned short* __restrict__ Vo,
    int M, int N, int K) {
  __shared__ unsigned short Alds[2][128][40];   // 80B row stride, 16B aligned
  __shared__ unsigned short Blds[2][128][40];

  const int tid  = threadIdx.x;
  const int lane = tid & 31;
  const int wave = tid >> 5;
  const int wm = wave >> 2;   // 0..1
  const int wn = wave & 3;    // 0..3
  const int n0 = blockIdx.x * 128;
  const int m0 = blockIdx.y * 128;

  v8f acc[4][2];
#pragma unroll
  for (int i = 0; i < 4; ++i)
#pragma unroll
    for (int j = 0; j < 2; ++j) acc[i][j] = {};

  const int kIters = K >> 5;

  // prologue: fill buffer 0
  stage_a_async(Abf, Alds[0], m0, K, 0, tid);
  stage_b_convert(Wg, Blds[0], n0, K, 0, tid);
  async_wait0();
  __syncthreads();

  for (int kt = 0; kt < kIters; ++kt) {
    const int cur = kt & 1;
    // prefetch tile kt+1 into the other buffer (async A copy overlaps WMMAs)
    if (kt + 1 < kIters)
      stage_a_async(Abf, Alds[cur ^ 1], m0, K, (kt + 1) << 5, tid);

    // ---- compute from buffer `cur` ----
    FragAB af[4], bfr[2];
#pragma unroll
    for (int mt = 0; mt < 4; ++mt)
      load_frag_a(af[mt], &Alds[cur][wm * 64 + mt * 16][0], 40, lane);
#pragma unroll
    for (int nt = 0; nt < 2; ++nt)
      load_frag_b(bfr[nt], &Blds[cur][wn * 32 + nt * 16][0], 40, lane);
#pragma unroll
    for (int mt = 0; mt < 4; ++mt)
#pragma unroll
      for (int nt = 0; nt < 2; ++nt)
        acc[mt][nt] = wmma_bf16(af[mt], bfr[nt], acc[mt][nt]);

    // stage next W tile (convert path) after compute issue
    if (kt + 1 < kIters)
      stage_b_convert(Wg, Blds[cur ^ 1], n0, K, (kt + 1) << 5, tid);
    // L2 prefetch of the tile after next (global_prefetch_b8)
    if (kt + 2 < kIters && tid < 128)
      __builtin_prefetch(Wg + (size_t)(n0 + tid) * K + ((kt + 2) << 5), 0, 1);

    async_wait0();
    __syncthreads();
  }

  // ---- epilogue ----
  const int half = lane >> 4;
  const int nlan = lane & 15;
#pragma unroll
  for (int mt = 0; mt < 4; ++mt) {
#pragma unroll
    for (int nt = 0; nt < 2; ++nt) {
#pragma unroll
      for (int r = 0; r < 8; ++r) {
        float val = acc[mt][nt][r];
        int grow = m0 + wm * 64 + mt * 16 + r + 8 * half;   // C: M = r (+8 upper half)
        int gcol = n0 + wn * 32 + nt * 16 + nlan;           // C: N = lane%16
        if constexpr (SCATTER_QKV) {
          int which = gcol >> 12;      // 0:q 1:k 2:v (H = 4096)
          int hd = gcol & 4095;
          int h = hd >> 7, d = hd & 127;
          int b = grow >> 11, s = grow & 2047;               // S = 2048
          unsigned short* dst = (which == 0) ? Qo : (which == 1) ? Ko : Vo;
          dst[((((size_t)b * NHh + h) * Ss + s) << 7) + d] = f2bf(val);
        } else {
          Cout[(size_t)grow * N + gcol] = val;
        }
      }
    }
  }
}

// ---------------------------------------------------------------------------
// RoPE on bf16 Q,K (head-major [b][h][s][d]); folds 1/sqrt(HD) into Q.
// ---------------------------------------------------------------------------
__global__ __launch_bounds__(256) void rope_kernel(unsigned short* __restrict__ Q,
                                                   unsigned short* __restrict__ Kd,
                                                   const int* __restrict__ pos) {
  const float scale = 0.08838834764831845f;   // 128^-0.5
  int idx = blockIdx.x * blockDim.x + threadIdx.x;  // B*NH*S*64 threads
  int d  = idx & 63;
  int s  = (idx >> 6) & (Ss - 1);
  int bh = idx >> 17;                                // 64*2048 = 2^17
  int b  = bh >> 5;
  int p  = pos[b * Ss + s];
  float inv = __powf(10000.0f, -(float)d * (1.0f / 64.0f));
  float ang = (float)p * inv;
  float sn, c;
  __sincosf(ang, &sn, &c);
  size_t base = ((size_t)bh * Ss + s) << 7;          // *HD
  float q1 = bf2f(Q[base + d]), q2 = bf2f(Q[base + d + 64]);
  Q[base + d]      = f2bf((q1 * c - q2 * sn) * scale);
  Q[base + d + 64] = f2bf((q2 * c + q1 * sn) * scale);
  float k1 = bf2f(Kd[base + d]), k2 = bf2f(Kd[base + d + 64]);
  Kd[base + d]      = f2bf(k1 * c - k2 * sn);
  Kd[base + d + 64] = f2bf(k2 * c + k1 * sn);
}

// ---------------------------------------------------------------------------
// Causal flash attention. Grid (S/64, B*NH), 128 threads (4 waves).
// Q in registers; K block async-copied to LDS (bf16, layout already matches
// the B operand); V transposed into LDS; P round-trips via wave-private LDS.
// ---------------------------------------------------------------------------
__global__ __launch_bounds__(128) void attn_kernel(const unsigned short* __restrict__ Q,
                                                   const unsigned short* __restrict__ Kg,
                                                   const unsigned short* __restrict__ Vg,
                                                   unsigned short* __restrict__ Og) {
  __shared__ unsigned short Klds[64][136];    // K block: [key][dim], padded
  __shared__ unsigned short Vt[128][72];      // V^T: [dim][key], padded
  __shared__ unsigned short Pl[4][16][72];    // per-wave P tile: [row][key]

  const int tid  = threadIdx.x;
  const int lane = tid & 31;
  const int wave = tid >> 5;
  const int half = lane >> 4;
  const int nlan = lane & 15;

  const int qblk = blockIdx.x;
  const int bh   = blockIdx.y;
  const int b = bh >> 5;
  const int h = bh & 31;
  const int q0  = qblk * 64;
  const int qr0 = q0 + wave * 16;

  const unsigned short* qp = Q  + (size_t)bh * Ss * HDd;
  const unsigned short* kp = Kg + (size_t)bh * Ss * HDd;
  const unsigned short* vp = Vg + (size_t)bh * Ss * HDd;

  FragAB qf[4];
#pragma unroll
  for (int kk = 0; kk < 4; ++kk)
    load_frag_a(qf[kk], qp + (size_t)qr0 * HDd + kk * 32, HDd, lane);

  v8f o[8];
#pragma unroll
  for (int t = 0; t < 8; ++t) o[t] = {};
  float rm[8], rl[8];
#pragma unroll
  for (int r = 0; r < 8; ++r) { rm[r] = -1e30f; rl[r] = 0.0f; }

  for (int jb = 0; jb <= qblk; ++jb) {
    const int j0 = jb * 64;
    __syncthreads();                 // all waves done reading previous K/Vt
    // ---- stage K block via async global->LDS (no transform needed) ----
#pragma unroll
    for (int i = 0; i < 8; ++i) {
      int c   = tid + i * 128;       // 1024 chunks of 8 elems
      int row = c >> 4;              // key 0..63
      int d0  = (c & 15) << 3;       // dim 0,8,..,120
      async_copy_b128(kp + (size_t)(j0 + row) * HDd + d0, &Klds[row][d0]);
    }
    // ---- stage V^T into LDS (transpose: global b128 -> scattered b16) ----
#pragma unroll
    for (int i = 0; i < 8; ++i) {
      int c   = tid + i * 128;
      int row = c >> 4;
      int d0  = (c & 15) << 3;
      u32x4 v = *(const u32x4*)(vp + (size_t)(j0 + row) * HDd + d0);
      const unsigned short* ve = (const unsigned short*)&v;
#pragma unroll
      for (int j = 0; j < 8; ++j) Vt[d0 + j][row] = ve[j];
    }
    async_wait0();
    __syncthreads();

    // ---- scores: S = Q @ K^T (K B-frags from LDS) ----
    v8f sc[4];
#pragma unroll
    for (int nt = 0; nt < 4; ++nt) sc[nt] = {};
#pragma unroll
    for (int nt = 0; nt < 4; ++nt) {
#pragma unroll
      for (int kk = 0; kk < 4; ++kk) {
        FragAB kb;
        load_frag_b(kb, &Klds[nt * 16][kk * 32], 136, lane);
        sc[nt] = wmma_bf16(qf[kk], kb, sc[nt]);
      }
    }
    // ---- causal mask on diagonal block ----
    if (jb == qblk) {
#pragma unroll
      for (int nt = 0; nt < 4; ++nt) {
        int col = j0 + nt * 16 + nlan;
#pragma unroll
        for (int r = 0; r < 8; ++r) {
          int row = qr0 + r + 8 * half;
          if (col > row) sc[nt][r] = -1e30f;
        }
      }
    }
    // ---- online softmax (row stats live per 16-lane half) ----
    float mnew[8];
#pragma unroll
    for (int r = 0; r < 8; ++r) mnew[r] = rm[r];
#pragma unroll
    for (int nt = 0; nt < 4; ++nt) {
#pragma unroll
      for (int r = 0; r < 8; ++r) {
        float v = sc[nt][r];
        v = fmaxf(v, __shfl_xor(v, 1, 32));
        v = fmaxf(v, __shfl_xor(v, 2, 32));
        v = fmaxf(v, __shfl_xor(v, 4, 32));
        v = fmaxf(v, __shfl_xor(v, 8, 32));
        mnew[r] = fmaxf(mnew[r], v);
      }
    }
#pragma unroll
    for (int r = 0; r < 8; ++r) {
      float f = __expf(rm[r] - mnew[r]);
      rl[r] *= f;
#pragma unroll
      for (int t = 0; t < 8; ++t) o[t][r] *= f;
      rm[r] = mnew[r];
    }
#pragma unroll
    for (int nt = 0; nt < 4; ++nt) {
#pragma unroll
      for (int r = 0; r < 8; ++r) {
        float p = __expf(sc[nt][r] - rm[r]);
        float sum = p;
        sum += __shfl_xor(sum, 1, 32);
        sum += __shfl_xor(sum, 2, 32);
        sum += __shfl_xor(sum, 4, 32);
        sum += __shfl_xor(sum, 8, 32);
        rl[r] += sum;
        Pl[wave][r + 8 * half][nt * 16 + nlan] = f2bf(p);
      }
    }
    // ---- O += P @ V  (wave-private LDS; ds ops in-order per wave) ----
#pragma unroll
    for (int k2 = 0; k2 < 2; ++k2) {
      FragAB pa;
      load_frag_a(pa, &Pl[wave][0][k2 * 32], 72, lane);
#pragma unroll
      for (int dt = 0; dt < 8; ++dt) {
        FragAB vb;
        load_frag_b(vb, &Vt[dt * 16][k2 * 32], 72, lane);
        o[dt] = wmma_bf16(pa, vb, o[dt]);
      }
    }
  }

  // ---- normalize + store bf16 attn-out in [B][S][H] layout ----
#pragma unroll
  for (int dt = 0; dt < 8; ++dt) {
#pragma unroll
    for (int r = 0; r < 8; ++r) {
      int s = qr0 + r + 8 * half;
      int d = dt * 16 + nlan;
      float val = o[dt][r] / rl[r];
      Og[((size_t)b * Ss + s) * Hh + h * HDd + d] = f2bf(val);
    }
  }
}

// ---------------------------------------------------------------------------
extern "C" void kernel_launch(void* const* d_in, const int* in_sizes, int n_in,
                              void* d_out, int out_size, void* d_ws, size_t ws_size,
                              hipStream_t stream) {
  (void)in_sizes; (void)n_in; (void)out_size; (void)ws_size;
  const float* hidden    = (const float*)d_in[0];
  const int*   positions = (const int*)d_in[1];
  const float* w_pack    = (const float*)d_in[2];
  const float* w_o       = (const float*)d_in[3];

  char* ws = (char*)d_ws;
  const size_t planeBytes = (size_t)Bb * NHh * Ss * HDd * sizeof(unsigned short); // 32 MiB
  unsigned short* hb = (unsigned short*)(ws);                   // hidden bf16
  unsigned short* qb = (unsigned short*)(ws + planeBytes);
  unsigned short* kb = (unsigned short*)(ws + 2 * planeBytes);
  unsigned short* vb = (unsigned short*)(ws + 3 * planeBytes);
  unsigned short* ob = (unsigned short*)(ws + 4 * planeBytes);  // attn out bf16

  // 0) hidden fp32 -> bf16 (one pass; enables async A staging in both GEMMs)
  int nElems = Bb * Ss * Hh;                    // 16.78M
  cvt_kernel<<<dim3(nElems / (256 * 4)), dim3(256), 0, stream>>>(hidden, hb);

  // 1) QKV projection, scatter to head-major bf16 q/k/v
  dim3 g1((3 * Hh) / 128, (Bb * Ss) / 128);
  gemm_wmma<true><<<g1, dim3(256), 0, stream>>>(
      hb, w_pack, (float*)nullptr, qb, kb, vb, Bb * Ss, 3 * Hh, Hh);

  // 2) RoPE (1/sqrt(HD) folded into Q)
  int ropeThreads = Bb * NHh * Ss * 64;
  rope_kernel<<<dim3(ropeThreads / 256), dim3(256), 0, stream>>>(qb, kb, positions);

  // 3) causal flash attention
  attn_kernel<<<dim3(Ss / 64, Bb * NHh), dim3(128), 0, stream>>>(qb, kb, vb, ob);

  // 4) output projection -> fp32 d_out
  dim3 g2(Hh / 128, (Bb * Ss) / 128);
  gemm_wmma<false><<<g2, dim3(256), 0, stream>>>(
      ob, w_o, (float*)d_out,
      (unsigned short*)nullptr, (unsigned short*)nullptr, (unsigned short*)nullptr,
      Bb * Ss, Hh, Hh);
}
